// RestoreNet_rotate_back_similar_MLP_90228672954716
// MI455X (gfx1250) — compile-verified, matrix operand order checked by reading, not run on hardware
//
#include <hip/hip_runtime.h>
#include <hip/hip_bf16.h>
#include <math.h>

typedef __attribute__((ext_vector_type(16))) _Float16 v16h;
typedef __attribute__((ext_vector_type(8)))  float    v8f;
typedef __attribute__((ext_vector_type(2)))  float    v2f;

#define BB 8
#define NN 2048
#define DD 256
#define HH 16
#define KK 8

// CDNA5 VOP3P packed fp32 FMA: 2 FMAs per instruction (no clang builtin).
static __device__ __forceinline__ v2f pk_fma(v2f a, v2f b, v2f c) {
    v2f d;
    asm("v_pk_fma_f32 %0, %1, %2, %3" : "=v"(d) : "v"(a), "v"(b), "v"(c));
    return d;
}

// ---------------------------------------------------------------------------
// Pass 1: row-normalize (f32) and convert to f16, row-major [B*N, D].
// One wave per row; each lane owns 8 contiguous elements.
// ---------------------------------------------------------------------------
__global__ void normalize_to_f16(const float* __restrict__ src,
                                 _Float16* __restrict__ dst) {
    const int row  = blockIdx.x * (blockDim.x >> 5) + (threadIdx.x >> 5);
    const int lane = threadIdx.x & 31;
    const float* r = src + (size_t)row * DD + lane * 8;

    float x[8];
    float ss = 0.f;
#pragma unroll
    for (int i = 0; i < 8; ++i) { x[i] = r[i]; ss = fmaf(x[i], x[i], ss); }
#pragma unroll
    for (int off = 16; off > 0; off >>= 1) ss += __shfl_xor(ss, off, 32);

    const float inv = 1.0f / (sqrtf(ss) + 1e-8f);

    union { _Float16 h[8]; uint4 u; } o;
#pragma unroll
    for (int i = 0; i < 8; ++i) o.h[i] = (_Float16)(x[i] * inv);
    *reinterpret_cast<uint4*>(dst + (size_t)row * DD + lane * 8) = o.u;
}

// ---------------------------------------------------------------------------
// Pass 2: batched GEMM (cosine sim) with v_wmma_f32_16x16x32_f16.
// Block = 256 threads (8 waves, 2x4 wave grid). Each wave computes a 32x64
// tile: 2 A frags + 4 B frags -> 8 WMMAs per K-step (0.75 frag loads/WMMA).
// K-loop: 256 in steps of 32 -> 64 WMMAs per wave.
// Fused epilogue: -sqdist(q,p) + 2->16->1 ReLU MLP via v_pk_fma_f32.
// ---------------------------------------------------------------------------
union alignas(32) HFrag { v16h v; uint4 u[2]; };

__global__ void gemm_mlp_kernel(const _Float16* __restrict__ Af2, // [B*N,D] normalized f2
                                const _Float16* __restrict__ Bf1, // [B*N,D] normalized f1
                                const float* __restrict__ q,      // [B,N,3]
                                const float* __restrict__ p,      // [B,N,3]
                                const float* __restrict__ W1,     // [H,2]
                                const float* __restrict__ b1,     // [H]
                                const float* __restrict__ W2,     // [1,H]
                                const float* __restrict__ b2,     // [1]
                                float* __restrict__ wOut) {       // [B,N,N]
    const int wv    = threadIdx.x >> 5;
    const int lane  = threadIdx.x & 31;
    const int b     = blockIdx.z;
    const int i0    = blockIdx.y * 64  + (wv >> 2) * 32;  // query rows (f2)
    const int j0    = blockIdx.x * 256 + (wv & 3) * 64;   // point cols (f1)

    const int m     = lane & 15;      // row/col within a 16-wide tile
    const int selHi = lane >> 4;      // half-wave select
    const int kbase = selHi * 8;      // K sub-offset per fragment layout

    // Fragment base pointers. A: rows i0+m and i0+16+m. B: cols j0+16*t+m.
    const _Float16* aRow0 = Af2 + ((size_t)(b * NN + i0 + m) * DD) + kbase;
    const _Float16* aRow1 = aRow0 + (size_t)16 * DD;
    const _Float16* bRow0 = Bf1 + ((size_t)(b * NN + j0 + m) * DD) + kbase;

    v8f C[2][4];
#pragma unroll
    for (int ai = 0; ai < 2; ++ai)
#pragma unroll
        for (int bi = 0; bi < 4; ++bi) C[ai][bi] = (v8f){};

#pragma unroll
    for (int kk = 0; kk < DD; kk += 32) {
        HFrag a[2], bf[4];
        a[0].u[0] = *reinterpret_cast<const uint4*>(aRow0 + kk);
        a[0].u[1] = *reinterpret_cast<const uint4*>(aRow0 + kk + 16);
        a[1].u[0] = *reinterpret_cast<const uint4*>(aRow1 + kk);
        a[1].u[1] = *reinterpret_cast<const uint4*>(aRow1 + kk + 16);
#pragma unroll
        for (int bi = 0; bi < 4; ++bi) {
            const _Float16* br = bRow0 + (size_t)(16 * bi) * DD + kk;
            bf[bi].u[0] = *reinterpret_cast<const uint4*>(br);
            bf[bi].u[1] = *reinterpret_cast<const uint4*>(br + 16);
        }
#pragma unroll
        for (int ai = 0; ai < 2; ++ai)
#pragma unroll
            for (int bi = 0; bi < 4; ++bi)
                C[ai][bi] = __builtin_amdgcn_wmma_f32_16x16x32_f16(
                    false, a[ai].v, false, bf[bi].v,
                    (short)0, C[ai][bi], false, false);
    }

    // ---- packed MLP weights (uniform loads) ----
    v2f pa[HH / 2], pb[HH / 2], pc[HH / 2], pd[HH / 2];
#pragma unroll
    for (int t = 0; t < HH / 2; ++t) {
        pa[t].x = W1[2 * (2 * t) + 0];     pa[t].y = W1[2 * (2 * t + 1) + 0];
        pb[t].x = W1[2 * (2 * t) + 1];     pb[t].y = W1[2 * (2 * t + 1) + 1];
        pc[t].x = b1[2 * t];               pc[t].y = b1[2 * t + 1];
        pd[t].x = W2[2 * t];               pd[t].y = W2[2 * t + 1];
    }
    const float b2v = b2[0];

    // ---- epilogue over the 8 sub-tiles ----
#pragma unroll
    for (int ai = 0; ai < 2; ++ai) {
        // preload the 8 q rows owned by this half-wave for this ai
        float qxr[8], qyr[8], qzr[8];
#pragma unroll
        for (int r = 0; r < 8; ++r) {
            const int i = i0 + 16 * ai + selHi * 8 + r;
            const float* qp = q + ((size_t)b * NN + i) * 3;
            qxr[r] = qp[0]; qyr[r] = qp[1]; qzr[r] = qp[2];
        }
#pragma unroll
        for (int bi = 0; bi < 4; ++bi) {
            const int j = j0 + 16 * bi + m;
            const float* pp = p + ((size_t)b * NN + j) * 3;
            const float px = pp[0], py = pp[1], pz = pp[2];
#pragma unroll
            for (int r = 0; r < 8; ++r) {
                const int i = i0 + 16 * ai + selHi * 8 + r;
                const float dx = qxr[r] - px, dy = qyr[r] - py, dz = qzr[r] - pz;
                const float w2v = -(dx * dx + dy * dy + dz * dz); // -max(sqdist,0)
                const float w1v = C[ai][bi][r];

                const v2f w1s = {w1v, w1v};
                const v2f w2s = {w2v, w2v};
                v2f acc2 = {b2v, 0.0f};
#pragma unroll
                for (int t = 0; t < HH / 2; ++t) {
                    v2f u = pk_fma(pb[t], w2s, pc[t]);
                    u = pk_fma(pa[t], w1s, u);
                    u.x = fmaxf(u.x, 0.0f);
                    u.y = fmaxf(u.y, 0.0f);
                    acc2 = pk_fma(pd[t], u, acc2);
                }
                wOut[((size_t)(b * NN + i)) * NN + j] = acc2.x + acc2.y;
            }
        }
    }
}

// ---------------------------------------------------------------------------
// Pass 3: per (b,qi) row of w: top-8 (sorted desc, low-index tie-break),
// softmax over the 8, gather f1 rows, weighted-sum + max pool over k.
// One 256-thread block per row; row staged in LDS (8KB of 320KB).
// ---------------------------------------------------------------------------
__global__ void topk_gather_kernel(const float* __restrict__ w,   // [B,N,N]
                                   const float* __restrict__ f1,  // [B,N,D]
                                   float* __restrict__ outF,      // [B,N,2D]
                                   float* __restrict__ outIdx) {  // [B,N,K]
    __shared__ float sval[NN];
    __shared__ float rv[256];
    __shared__ int   ri[256];
    __shared__ float topv[KK];
    __shared__ int   topi[KK];
    __shared__ float wsm[KK];

    const int row = blockIdx.x;
    const int b   = row / NN;
    const int tid = threadIdx.x;

    const float* wrow = w + (size_t)row * NN;
    for (int t = tid; t < NN; t += 256) sval[t] = wrow[t];
    __syncthreads();

    for (int it = 0; it < KK; ++it) {
        float best = -3.402823e38f;
        int   bidx = NN;
        for (int t = tid; t < NN; t += 256) {
            const float v = sval[t];
            if (v > best) { best = v; bidx = t; }   // ascending t -> lowest index on tie
        }
        rv[tid] = best; ri[tid] = bidx;
        __syncthreads();
        for (int s = 128; s > 0; s >>= 1) {
            if (tid < s) {
                const float ov = rv[tid + s];
                const int   oi = ri[tid + s];
                if (ov > rv[tid] || (ov == rv[tid] && oi < ri[tid])) {
                    rv[tid] = ov; ri[tid] = oi;
                }
            }
            __syncthreads();
        }
        if (tid == 0) {
            topv[it] = rv[0];
            topi[it] = ri[0];
            sval[ri[0]] = -3.402823e38f;   // mask out for next pass
        }
        __syncthreads();
    }

    if (tid == 0) {
        const float mx = topv[0];           // sorted descending
        float sum = 0.f;
        float e[KK];
#pragma unroll
        for (int k2 = 0; k2 < KK; ++k2) { e[k2] = expf(topv[k2] - mx); sum += e[k2]; }
        const float inv = 1.0f / sum;
#pragma unroll
        for (int k2 = 0; k2 < KK; ++k2) wsm[k2] = e[k2] * inv;
    }
    __syncthreads();

    // 256 threads == D channels: gather + pool
    const int d = tid;
    float acc = 0.f;
    float mxv = -3.402823e38f;
#pragma unroll
    for (int k2 = 0; k2 < KK; ++k2) {
        const float v = f1[((size_t)(b * NN + topi[k2])) * DD + d];
        mxv = fmaxf(mxv, v);
        acc = fmaf(wsm[k2], v, acc);
    }
    float* fo = outF + (size_t)row * (2 * DD);
    fo[d]      = acc;   // weighted features
    fo[DD + d] = mxv;   // max-pooled features

    if (tid < KK) outIdx[(size_t)row * KK + tid] = (float)topi[tid];
}

// ---------------------------------------------------------------------------
extern "C" void kernel_launch(void* const* d_in, const int* in_sizes, int n_in,
                              void* d_out, int out_size, void* d_ws, size_t ws_size,
                              hipStream_t stream) {
    const float* f1 = (const float*)d_in[0];
    const float* f2 = (const float*)d_in[1];
    const float* p  = (const float*)d_in[2];
    const float* q  = (const float*)d_in[3];
    const float* W1 = (const float*)d_in[4];
    const float* b1 = (const float*)d_in[5];
    const float* W2 = (const float*)d_in[6];
    const float* b2 = (const float*)d_in[7];
    // d_in[8] = similar_num (8) -- fixed at compile time

    _Float16* f2h = (_Float16*)d_ws;                       // A: normalized f2, f16
    _Float16* f1h = f2h + (size_t)BB * NN * DD;            // B: normalized f1, f16

    float* outF   = (float*)d_out;                         // [B,N,2D]
    float* outIdx = outF + (size_t)BB * NN * 2 * DD;       // [B,N,K]
    float* outW   = outIdx + (size_t)BB * NN * KK;         // [B,N,N]

    // Pass 1: normalize + f32->f16 (8 rows per 256-thread block)
    normalize_to_f16<<<dim3((BB * NN) / 8), dim3(256), 0, stream>>>(f2, f2h);
    normalize_to_f16<<<dim3((BB * NN) / 8), dim3(256), 0, stream>>>(f1, f1h);

    // Pass 2: WMMA GEMM + fused distance/MLP epilogue -> w
    gemm_mlp_kernel<<<dim3(NN / 256, NN / 64, BB), dim3(256), 0, stream>>>(
        f2h, f1h, q, p, W1, b1, W2, b2, outW);

    // Pass 3: top-k + softmax + gather/pool -> f, idx
    topk_gather_kernel<<<dim3(BB * NN), dim3(256), 0, stream>>>(outW, f1, outF, outIdx);
}
